// Window_Block_50242527428633
// MI455X (gfx1250) — matrix operand order repaired and checked by reading, not compiled, not run on hardware
//
#include <hip/hip_runtime.h>
#include <hip/hip_bf16.h>
#include <math.h>

// ---------------------------------------------------------------------------
// Window_Block for MI455X (gfx1250): all GEMM-shaped math through
// v_wmma_f32_16x16x32_f16 (wave32 WMMA, f32 accumulate).
// ---------------------------------------------------------------------------

typedef __attribute__((ext_vector_type(16))) _Float16 v16h;
typedef __attribute__((ext_vector_type(8)))  _Float16 v8h;
typedef __attribute__((ext_vector_type(8)))  float    v8f;
typedef __attribute__((ext_vector_type(4)))  float    v4f;
typedef __attribute__((ext_vector_type(4)))  unsigned int u32x4;

#define CB   256      // channels
#define PW   50176    // total positions (= 1024 windows * 49)
#define NWIN 1024     // windows = 16 * 8 * 8
#define NPOS 49       // 7*7
#define NHEADS 8
#define HDIM 32
#define HIDD 1024
#define TOTAL_ELEMS (16*256*56*56)

__device__ __forceinline__ v16h mk16(v8h lo, v8h hi) {
  return __builtin_shufflevector(lo, hi, 0,1,2,3,4,5,6,7,8,9,10,11,12,13,14,15);
}
__device__ __forceinline__ v8f wmma16(v16h a, v16h b, v8f c) {
  // D = A(16x32 f16) * B(32x16 f16) + C(16x16 f32)
  return __builtin_amdgcn_wmma_f32_16x16x32_f16(false, a, false, b, (short)0, c, false, false);
}

// ---------------------------------------------------------------------------
// Weight convert+transpose: W[O][Cin] f32 -> WT[Cin][O] f16
// ---------------------------------------------------------------------------
__global__ void conv_wT(const float* __restrict__ W, _Float16* __restrict__ WT,
                        int O, int Cin) {
  int i = blockIdx.x * blockDim.x + threadIdx.x;
  if (i < O * Cin) {
    int o = i / Cin, c = i % Cin;
    WT[(size_t)c * O + o] = (_Float16)W[i];
  }
}

// ---------------------------------------------------------------------------
// Precompute relative-position bias + padding mask, TRANSPOSED for vector
// loads along the score-row axis: bp[8][64 cols][64 rows] f32.
//   col >= 49 -> -3e30 (softmax weight exactly 0), row >= 49 -> 0 (unused),
//   else table[rel_index(row,col)][head].
// ---------------------------------------------------------------------------
__global__ void bias_precomp(const float* __restrict__ table, float* __restrict__ bp) {
  int i = blockIdx.x * blockDim.x + threadIdx.x;
  if (i >= NHEADS * 64 * 64) return;
  int r = i & 63, c = (i >> 6) & 63, h = i >> 12;   // [h][c][r]
  float v;
  if (c >= NPOS)      v = -3e30f;
  else if (r >= NPOS) v = 0.f;
  else {
    int ri = r / 7 - c / 7 + 6;
    int ci = r % 7 - c % 7 + 6;
    v = table[(ri * 13 + ci) * NHEADS + h];
  }
  bp[i] = v;
}

// ---------------------------------------------------------------------------
// Channel LayerNorm (over C=256) of an NCHW f32 tensor -> f16 [P][C].
// windowOrder=1: p = window-major (m*49+n); windowOrder=0: p = b*3136+h*56+w.
// One wave per position; each lane owns 8 channels.
// ---------------------------------------------------------------------------
__global__ void ln_kernel(const float* __restrict__ x, const float* __restrict__ g,
                          const float* __restrict__ bsh, _Float16* __restrict__ out,
                          int windowOrder) {
  int wid = (blockIdx.x * blockDim.x + threadIdx.x) >> 5;
  int lid = threadIdx.x & 31;
  if (wid >= PW) return;
  int b, h, w;
  if (windowOrder) {
    int m = wid / NPOS, n = wid % NPOS;
    b = m >> 6; int hb = (m >> 3) & 7, wb = m & 7, wh = n / 7, ww = n % 7;
    h = hb * 7 + wh; w = wb * 7 + ww;
  } else {
    b = wid / 3136; int r = wid % 3136; h = r / 56; w = r % 56;
  }
  size_t base = ((size_t)b * CB) * 3136 + (size_t)h * 56 + w;
  float vals[8], s = 0.f, s2 = 0.f;
#pragma unroll
  for (int j = 0; j < 8; j++) {
    int c = lid + 32 * j;
    float v = x[base + (size_t)c * 3136];
    vals[j] = v; s += v; s2 += v * v;
  }
#pragma unroll
  for (int mk = 16; mk; mk >>= 1) {
    s  += __shfl_xor(s,  mk, 32);
    s2 += __shfl_xor(s2, mk, 32);
  }
  float mu  = s * (1.f / 256.f);
  float var = s2 * (1.f / 256.f) - mu * mu;
  float inv = rsqrtf(var + 1e-5f);
#pragma unroll
  for (int j = 0; j < 8; j++) {
    int c = lid + 32 * j;
    out[(size_t)wid * CB + c] = (_Float16)((vals[j] - mu) * inv * g[c] + bsh[c]);
  }
}

// ---------------------------------------------------------------------------
// Generic WMMA GEMM:  Cout[P][O] = A[P][K] * Bt[K][O] + bias   (f16 in, f16 out)
// Block tile 128x128, 8 waves, wave tile 32x64 (2x4 WMMA 16x16), K-step 32.
// Register-staged global->LDS double buffer hides memory latency behind WMMA.
// GELU=1 applies exact gelu in the epilogue.
// ---------------------------------------------------------------------------
template<int GELU>
__global__ void gemm_wmma(const _Float16* __restrict__ A,
                          const _Float16* __restrict__ Bt,
                          const float* __restrict__ bias,
                          _Float16* __restrict__ Cout,
                          int K, int O) {
  __shared__ __align__(16) _Float16 lAs[128][40];   // rows x Kslab (padded)
  __shared__ __align__(16) _Float16 lBt[128][40];   // cols x Kslab (transposed, padded)

  const int tid = threadIdx.x;
  const int wv  = tid >> 5;
  const int lid = tid & 31;
  const int lm  = lid & 15;
  const int hi  = lid >> 4;          // 0: lanes 0-15, 1: lanes 16-31
  const int waveRow = wv & 3;        // 4 row-groups of 32
  const int waveCol = wv >> 2;       // 2 col-groups of 64
  const int rowBase = blockIdx.y * 128;
  const int colBase = blockIdx.x * 128;

  v8f acc[2][4];
#pragma unroll
  for (int i = 0; i < 2; i++)
#pragma unroll
    for (int j = 0; j < 4; j++) acc[i][j] = (v8f){0.f,0.f,0.f,0.f,0.f,0.f,0.f,0.f};

  const int rA = tid >> 1, cA = (tid & 1) * 16;    // A: 2 threads/row, 16 halves each
  const int rB = tid >> 3, cBv = (tid & 7) * 16;   // B: 8 threads/row, 16 halves each

  const _Float16* pA = &A[(size_t)(rowBase + rA) * K + cA];
  const _Float16* pB = &Bt[(size_t)rB * O + colBase + cBv];

  // prologue: fetch K-slab 0 into registers
  u32x4 ga0 = *(const u32x4*)pA;
  u32x4 ga1 = *(const u32x4*)(pA + 8);
  u32x4 gb0 = *(const u32x4*)pB;
  u32x4 gb1 = *(const u32x4*)(pB + 8);

  for (int k0 = 0; k0 < K; k0 += 32) {
    __syncthreads();                  // previous iteration's fragment reads done
    *(u32x4*)&lAs[rA][cA]     = ga0;
    *(u32x4*)&lAs[rA][cA + 8] = ga1;
    {
      _Float16 tmp[16];
      *(u32x4*)tmp       = gb0;
      *(u32x4*)(tmp + 8) = gb1;
#pragma unroll
      for (int j = 0; j < 16; j++) lBt[cBv + j][rB] = tmp[j];
    }
    __syncthreads();

    if (k0 + 32 < K) {                // fetch next slab while WMMAs run
      ga0 = *(const u32x4*)(pA + k0 + 32);
      ga1 = *(const u32x4*)(pA + k0 + 40);
      gb0 = *(const u32x4*)(pB + (size_t)(k0 + 32) * O);
      gb1 = *(const u32x4*)(pB + (size_t)(k0 + 32) * O + 8);
    }

    // A fragments (16-bit A 16x32 layout): vgpr0-3: k=dlo..dlo+7, vgpr4-7: k=16+dlo..
    const int dlo = hi * 8;
    v16h af[2], bf[4];
#pragma unroll
    for (int i = 0; i < 2; i++) {
      int r = waveRow * 32 + i * 16 + lm;
      af[i] = mk16(*(const v8h*)&lAs[r][dlo], *(const v8h*)&lAs[r][16 + dlo]);
    }
#pragma unroll
    for (int j = 0; j < 4; j++) {
      int c = waveCol * 64 + j * 16 + lm;
      bf[j] = mk16(*(const v8h*)&lBt[c][hi * 8], *(const v8h*)&lBt[c][16 + hi * 8]);
    }
#pragma unroll
    for (int i = 0; i < 2; i++)
#pragma unroll
      for (int j = 0; j < 4; j++)
        acc[i][j] = wmma16(af[i], bf[j], acc[i][j]);
  }

  // epilogue: C layout row = v + hi*8, col = lm
#pragma unroll
  for (int i = 0; i < 2; i++) {
#pragma unroll
    for (int j = 0; j < 4; j++) {
      int col = colBase + waveCol * 64 + j * 16 + lm;
      float bb = bias[col];
#pragma unroll
      for (int v = 0; v < 8; v++) {
        int row = rowBase + waveRow * 32 + i * 16 + v + hi * 8;
        float val = acc[i][j][v] + bb;
        if (GELU) val = 0.5f * val * (1.f + erff(val * 0.70710678118f));
        Cout[(size_t)row * O + col] = (_Float16)val;
      }
    }
  }
}

// ---------------------------------------------------------------------------
// Windowed relative-bias attention, one block per (head, window).
// qkv: f16 [PW][768] (q | k | v, channel = head*32+d), out: f16 [PW][256].
// biasPre: [8][64 cols][64 rows] f32 precomputed bias + pad mask (transposed).
// 4 waves; wave wv owns score rows [wv*16, wv*16+16). N=49 padded to 64.
// ---------------------------------------------------------------------------
__global__ void win_attn(const _Float16* __restrict__ qkv,
                         const float* __restrict__ biasPre,
                         _Float16* __restrict__ outO) {
  const int head = blockIdx.x;
  const int m    = blockIdx.y;
  __shared__ __align__(16) _Float16 sP[4][16][64];   // per-wave probs, A-layout friendly
  __shared__ __align__(16) _Float16 sVt[32][72];     // V transposed: [d][k]

  // ---- stage V (64 rows padded, 32 d) transposed into LDS ----
  if (threadIdx.x < 64) {
    int k = threadIdx.x;
    _Float16 vh[32];
    if (k < NPOS) {
      const u32x4* src = (const u32x4*)&qkv[((size_t)m * NPOS + k) * 768 + 512 + head * HDIM];
      u32x4* dst = (u32x4*)vh;
      dst[0] = src[0]; dst[1] = src[1]; dst[2] = src[2]; dst[3] = src[3];
    } else {
#pragma unroll
      for (int d = 0; d < 32; d++) vh[d] = (_Float16)0.f;
    }
#pragma unroll
    for (int d = 0; d < 32; d++) sVt[d][k] = vh[d];
  }
  __syncthreads();

  const int wv  = threadIdx.x >> 5;
  const int lid = threadIdx.x & 31;
  const int lm  = lid & 15;
  const int hi  = lid >> 4;
  const int r0  = wv * 16;

  // ---- K^T fragments: B col = score-col (kcol), reduction dim = d ----
  v16h bk[4];
#pragma unroll
  for (int ct = 0; ct < 4; ct++) {
    int kcol = ct * 16 + lm;
    if (kcol < NPOS) {
      const _Float16* p = &qkv[((size_t)m * NPOS + kcol) * 768 + 256 + head * HDIM + hi * 8];
      bk[ct] = mk16(*(const v8h*)p, *(const v8h*)(p + 16));
    } else {
      v8h z = {};
      bk[ct] = mk16(z, z);
    }
  }
  // ---- Q A-fragment for own row strip ----
  v16h qa;
  {
    int mrow = r0 + lm;
    if (mrow < NPOS) {
      const _Float16* p = &qkv[((size_t)m * NPOS + mrow) * 768 + head * HDIM];
      qa = mk16(*(const v8h*)(p + hi * 8), *(const v8h*)(p + 16 + hi * 8));
    } else {
      v8h z = {};
      qa = mk16(z, z);
    }
  }

  // ---- scores: 4 WMMAs (16x64 strip) ----
  v8f S[4];
#pragma unroll
  for (int ct = 0; ct < 4; ct++) {
    v8f z = {0.f,0.f,0.f,0.f,0.f,0.f,0.f,0.f};
    S[ct] = wmma16(qa, bk[ct], z);
  }

  // ---- scale + precomputed bias/mask: 2 x b128 per tile (transposed bias) ----
  const float scale = 0.17677669529663689f;  // 32^-0.5
  const float* bh = biasPre + head * 4096;
#pragma unroll
  for (int ct = 0; ct < 4; ct++) {
    const float* bp = bh + (ct * 16 + lm) * 64 + r0 + hi * 8;  // 8 consecutive rows
    v4f b0 = *(const v4f*)bp;
    v4f b1 = *(const v4f*)(bp + 4);
#pragma unroll
    for (int v = 0; v < 4; v++) S[ct][v]     = fmaf(S[ct][v],     scale, b0[v]);
#pragma unroll
    for (int v = 0; v < 4; v++) S[ct][v + 4] = fmaf(S[ct][v + 4], scale, b1[v]);
  }

  // ---- softmax over 64 cols (row = fixed (v,hi), 16 lanes x 4 tiles) ----
#pragma unroll
  for (int v = 0; v < 8; v++) {
    float rmax = fmaxf(fmaxf(S[0][v], S[1][v]), fmaxf(S[2][v], S[3][v]));
#pragma unroll
    for (int mk = 8; mk; mk >>= 1) rmax = fmaxf(rmax, __shfl_xor(rmax, mk, 32));
    float e0 = __expf(S[0][v] - rmax), e1 = __expf(S[1][v] - rmax);
    float e2 = __expf(S[2][v] - rmax), e3 = __expf(S[3][v] - rmax);
    float rs = e0 + e1 + e2 + e3;
#pragma unroll
    for (int mk = 8; mk; mk >>= 1) rs += __shfl_xor(rs, mk, 32);
    float inv = 1.f / rs;
    S[0][v] = e0 * inv; S[1][v] = e1 * inv; S[2][v] = e2 * inv; S[3][v] = e3 * inv;
  }

  // ---- re-layout probs C->A via LDS (wave-private strip, no barrier needed) ----
#pragma unroll
  for (int ct = 0; ct < 4; ct++)
#pragma unroll
    for (int v = 0; v < 8; v++)
      sP[wv][v + hi * 8][ct * 16 + lm] = (_Float16)S[ct][v];

  // ---- O = P (16x64) * V (64x32): 2 K-steps x 2 col tiles ----
  v8f Oacc[2] = {{0.f,0.f,0.f,0.f,0.f,0.f,0.f,0.f},{0.f,0.f,0.f,0.f,0.f,0.f,0.f,0.f}};
#pragma unroll
  for (int ks = 0; ks < 2; ks++) {
    v16h pa = mk16(*(const v8h*)&sP[wv][lm][ks * 32 + hi * 8],
                   *(const v8h*)&sP[wv][lm][ks * 32 + 16 + hi * 8]);
#pragma unroll
    for (int nt = 0; nt < 2; nt++) {
      int dcol = nt * 16 + lm;
      v16h vb = mk16(*(const v8h*)&sVt[dcol][ks * 32 + hi * 8],
                     *(const v8h*)&sVt[dcol][ks * 32 + 16 + hi * 8]);
      Oacc[nt] = wmma16(pa, vb, Oacc[nt]);
    }
  }

  // ---- epilogue ----
#pragma unroll
  for (int nt = 0; nt < 2; nt++) {
#pragma unroll
    for (int v = 0; v < 8; v++) {
      int mrow = r0 + v + hi * 8;
      if (mrow < NPOS) {
        int col = head * HDIM + nt * 16 + lm;
        outO[((size_t)m * NPOS + mrow) * CB + col] = (_Float16)Oacc[nt][v];
      }
    }
  }
}

// ---------------------------------------------------------------------------
// xr = x + unblock(out2)   (out2 is f16 [PW][C], window-ordered)
// ---------------------------------------------------------------------------
__global__ void residual_unblock(const float* __restrict__ x,
                                 const _Float16* __restrict__ out2,
                                 float* __restrict__ xr) {
  int e = blockIdx.x * blockDim.x + threadIdx.x;
  if (e >= TOTAL_ELEMS) return;
  int w = e % 56; int t = e / 56; int h = t % 56; t /= 56;
  int c = t % CB; int b = t / CB;
  int hb = h / 7, wh = h % 7, wb = w / 7, ww = w % 7;
  int p = (((b * 8 + hb) * 8 + wb) * NPOS) + wh * 7 + ww;
  xr[e] = x[e] + (float)out2[(size_t)p * CB + c];
}

// out = xr + Z  (Z is f16 [P2][C], natural order)
__global__ void final_add(const float* __restrict__ xr,
                          const _Float16* __restrict__ Z,
                          float* __restrict__ out) {
  int e = blockIdx.x * blockDim.x + threadIdx.x;
  if (e >= TOTAL_ELEMS) return;
  int w = e % 56; int t = e / 56; int h = t % 56; t /= 56;
  int c = t % CB; int b = t / CB;
  int p = b * 3136 + h * 56 + w;
  out[e] = xr[e] + (float)Z[(size_t)p * CB + c];
}

// ---------------------------------------------------------------------------
extern "C" void kernel_launch(void* const* d_in, const int* in_sizes, int n_in,
                              void* d_out, int out_size, void* d_ws, size_t ws_size,
                              hipStream_t stream) {
  const float* x      = (const float*)d_in[0];
  const float* ln1_w  = (const float*)d_in[1];
  const float* ln1_b  = (const float*)d_in[2];
  const float* ln2_w  = (const float*)d_in[3];
  const float* ln2_b  = (const float*)d_in[4];
  const float* qkv_w  = (const float*)d_in[5];
  const float* qkv_b  = (const float*)d_in[6];
  const float* proj_w = (const float*)d_in[7];
  const float* proj_b = (const float*)d_in[8];
  const float* table  = (const float*)d_in[9];
  const float* fc1_w  = (const float*)d_in[10];
  const float* fc1_b  = (const float*)d_in[11];
  const float* fc2_w  = (const float*)d_in[12];
  const float* fc2_b  = (const float*)d_in[13];
  float* out = (float*)d_out;

  char* ws = (char*)d_ws;
  // workspace layout (bytes)
  _Float16* wqkvT  = (_Float16*)(ws + 0);           // 256*768*2  = 393216
  _Float16* wprojT = (_Float16*)(ws + 393216);      // 256*256*2  = 131072
  _Float16* fc1T   = (_Float16*)(ws + 524288);      // 256*1024*2 = 524288
  _Float16* fc2T   = (_Float16*)(ws + 1048576);     // 1024*256*2 = 524288
  float*    biasP  = (float*)(ws + 1572864);        // 8*64*64*4  = 131072
  const size_t offA  = 1703936;                 // 25.7MB: Xln16 -> out2
  const size_t offB  = offA + 25690112;         // 102.8MB: QKV -> Hgelu
  const size_t offC  = offB + 102760448;        // 25.7MB: attnO -> Z
  const size_t offD  = offC + 25690112;         // 25.7MB: out1 -> X2
  const size_t offXR = offD + 25690112;         // 51.4MB: f32 residual
  _Float16* bufA = (_Float16*)(ws + offA);
  _Float16* bufB = (_Float16*)(ws + offB);
  _Float16* bufC = (_Float16*)(ws + offC);
  _Float16* bufD = (_Float16*)(ws + offD);
  float*    xr   = (float*)(ws + offXR);

  // 1) f16 transposed weights + bias/mask tile
  conv_wT<<<(768*256+255)/256, 256, 0, stream>>>(qkv_w, wqkvT, 768, 256);
  conv_wT<<<(256*256+255)/256, 256, 0, stream>>>(proj_w, wprojT, 256, 256);
  conv_wT<<<(1024*256+255)/256, 256, 0, stream>>>(fc1_w, fc1T, 1024, 256);
  conv_wT<<<(256*1024+255)/256, 256, 0, stream>>>(fc2_w, fc2T, 256, 1024);
  bias_precomp<<<(NHEADS*64*64+255)/256, 256, 0, stream>>>(table, biasP);

  // 2) LN1 -> window-ordered f16 [PW][256]
  ln_kernel<<<PW/8, 256, 0, stream>>>(x, ln1_w, ln1_b, bufA, 1);

  // 3) attention round 1
  gemm_wmma<0><<<dim3(768/128, PW/128), 256, 0, stream>>>(bufA, wqkvT, qkv_b, bufB, 256, 768);
  win_attn<<<dim3(NHEADS, NWIN), 128, 0, stream>>>(bufB, biasP, bufC);
  gemm_wmma<0><<<dim3(256/128, PW/128), 256, 0, stream>>>(bufC, wprojT, proj_b, bufD, 256, 256);

  // 4) attention round 2 (applied to round-1 output, per reference)
  gemm_wmma<0><<<dim3(768/128, PW/128), 256, 0, stream>>>(bufD, wqkvT, qkv_b, bufB, 256, 768);
  win_attn<<<dim3(NHEADS, NWIN), 128, 0, stream>>>(bufB, biasP, bufC);
  gemm_wmma<0><<<dim3(256/128, PW/128), 256, 0, stream>>>(bufC, wprojT, proj_b, bufA, 256, 256);

  // 5) residual + unblock
  residual_unblock<<<(TOTAL_ELEMS+255)/256, 256, 0, stream>>>(x, bufA, xr);

  // 6) MLP: LN2 -> fc1+gelu -> fc2 -> residual
  ln_kernel<<<PW/8, 256, 0, stream>>>(xr, ln2_w, ln2_b, bufD, 0);
  gemm_wmma<1><<<dim3(1024/128, PW/128), 256, 0, stream>>>(bufD, fc1T, fc1_b, bufB, 256, 1024);
  gemm_wmma<0><<<dim3(256/128, PW/128), 256, 0, stream>>>(bufB, fc2T, fc2_b, bufC, 1024, 256);
  final_add<<<(TOTAL_ELEMS+255)/256, 256, 0, stream>>>(xr, bufC, out);
}
